// ReceptorEncoder_19988777796166
// MI455X (gfx1250) — compile-verified, matrix operand order checked by reading, not run on hardware
//
#include <hip/hip_runtime.h>
#include <hip/hip_bf16.h>
#include <math.h>

// ===========================================================================
// ReceptorEncoder (EGNN x3 + keypoint attention pooling) for MI455X / gfx1250
//
// Fused per-edge MLP chain with bf16 WMMA (v_wmma_f32_16x16x32_bf16), f32
// accumulation, bf16 LDS-staged activation tiles (zero-convert A fragments),
// atomic scatter for segment sums. Weights pre-packed into per-lane WMMA
// B-fragments. Pooling folded to h @ v GEMV + softmax.
// ===========================================================================

typedef __attribute__((ext_vector_type(16))) __bf16 v16bf;
typedef __attribute__((ext_vector_type(8)))  float  v8f;

#define N_NODES 25600
#define N_GRAPHS 32
#define NPG 800
#define N_EDGES (25600 * 20)
#define KP 20

// bf16 LDS tile row stride: 136 bf16 = 272 B = 68 dwords; 68 % 64 == 4 so the
// 16 lanes of a half-wave (stride 4 banks) cover disjoint 4-bank groups.
static constexpr int TSTR = 136;

__device__ __forceinline__ float silu_f(float v) {
    // fast silu: v * rcp(1+exp(-v)); v_rcp_f32 is a TRANS op (co-executes w/ WMMA)
    return v * __builtin_amdgcn_rcpf(1.0f + __expf(-v));
}

// load 8 consecutive bf16 into A-fragment slots [base, base+8)
__device__ __forceinline__ void set8(v16bf& a, int base, const __bf16* __restrict__ p) {
#pragma unroll
    for (int j = 0; j < 8; ++j) a[base + j] = p[j];
}
// load 8 consecutive f32, convert to bf16, into A-fragment slots [base, base+8)
__device__ __forceinline__ void set8f(v16bf& a, int base, const float* __restrict__ p) {
#pragma unroll
    for (int j = 0; j < 8; ++j) a[base + j] = (__bf16)p[j];
}

__device__ __forceinline__ v8f wmma_bf16(v16bf a, v16bf b, v8f c) {
    return __builtin_amdgcn_wmma_f32_16x16x32_bf16(false, a, false, b, (short)0, c,
                                                   false, false);
}

// ---------------------------------------------------------------------------
// Weight fragment packing: W is row-major [K][N] (N==128 here).
// Fragment layout (CDNA5 16-bit B 32x16 layout, symmetric to A):
//   lane n (n<16): column N = nt*16+n, K runs {kc*32+0..7, kc*32+16..23}
//   lane n+16   : column N = nt*16+n, K runs {kc*32+8..15, kc*32+24..31}
// Stored as out[((nt*KC + kc)*32 + lane)*16 + j].
// ---------------------------------------------------------------------------
__global__ void pack_frags(const float* __restrict__ W, int K, int N,
                           __bf16* __restrict__ out) {
    int KC = K / 32;
    int total = (N / 16) * KC * 32 * 16;
    int i = blockIdx.x * blockDim.x + threadIdx.x;
    if (i >= total) return;
    int j = i & 15;
    int rest = i >> 4;
    int lane = rest & 31;
    rest >>= 5;
    int kc = rest % KC;
    int nt = rest / KC;
    int half = lane >> 4;
    int n = nt * 16 + (lane & 15);
    int k = kc * 32 + ((j < 8) ? (half * 8 + j) : (16 + half * 8 + (j - 8)));
    out[i] = (__bf16)W[(size_t)k * N + n];
}

__global__ void f2bf_kernel(const float* __restrict__ in, __bf16* __restrict__ out,
                            size_t n) {
    for (size_t i = blockIdx.x * (size_t)blockDim.x + threadIdx.x; i < n;
         i += (size_t)gridDim.x * blockDim.x)
        out[i] = (__bf16)in[i];
}

__global__ void copyf_kernel(const float* __restrict__ in, float* __restrict__ out,
                             size_t n) {
    for (size_t i = blockIdx.x * (size_t)blockDim.x + threadIdx.x; i < n;
         i += (size_t)gridDim.x * blockDim.x)
        out[i] = in[i];
}

__global__ void zero_kernel(float* __restrict__ p, size_t n) {
    for (size_t i = blockIdx.x * (size_t)blockDim.x + threadIdx.x; i < n;
         i += (size_t)gridDim.x * blockDim.x)
        p[i] = 0.0f;
}

// ---------------------------------------------------------------------------
// Fused edge kernel: 128 threads = 4 waves, each wave owns 16 edges.
//   m  = silu([h_s, h_d] @ eW1[0:2I] + radial*eW1[2I] + eb1)      (WMMA, K=2I)
//   msg= silu(m @ eW2 + eb2)                                       (WMMA, K=128)
//   cw = silu(msg @ cW1 + cb1) . cW2                               (WMMA + dot)
//   atomics: h_neigh[dst]+=msg ; x_sum[dst]+=cw*xdiff ; deg[dst]+=1
// ---------------------------------------------------------------------------
template <int INS>
__global__ __launch_bounds__(128) void edge_kernel(
    const float* __restrict__ xpos, const __bf16* __restrict__ hbf,
    const int* __restrict__ src, const int* __restrict__ dst,
    const __bf16* __restrict__ eW1f, const float* __restrict__ eW1rad,
    const float* __restrict__ eb1, const __bf16* __restrict__ eW2f,
    const float* __restrict__ eb2, const __bf16* __restrict__ cW1f,
    const float* __restrict__ cb1, const float* __restrict__ cW2,
    float* __restrict__ h_neigh, float* __restrict__ x_sum,
    float* __restrict__ deg) {
    constexpr int K1 = 2 * INS;
    constexpr int KC1 = K1 / 32;

    __shared__ __bf16 tA[4][16 * TSTR];
    __shared__ __bf16 tG[4][16 * TSTR];
    __shared__ float cwS[4][16];

    const int tid = threadIdx.x, wave = tid >> 5, lane = tid & 31;
    const int m = lane & 15, half = lane >> 4, m0 = half * 8;
    const int eBase = blockIdx.x * 64 + wave * 16;
    const int e = eBase + m;

    const int sIdx = src[e];
    const int dIdx = dst[e];

    // edge geometry (each half-wave computes its row's geometry redundantly)
    float dx = xpos[sIdx * 3 + 0] - xpos[dIdx * 3 + 0];
    float dy = xpos[sIdx * 3 + 1] - xpos[dIdx * 3 + 1];
    float dz = xpos[sIdx * 3 + 2] - xpos[dIdx * 3 + 2];
    float radial = dx * dx + dy * dy + dz * dz;
    float inv = 1.0f / (sqrtf(radial) + 1e-30f);
    float nx = dx * inv, ny = dy * inv, nz = dz * inv;

    // radial of rows m0..m0+7 (nt-invariant: hoisted, 8 bpermutes total)
    float radv[8];
#pragma unroll
    for (int r = 0; r < 8; ++r) radv[r] = __shfl(radial, m0 + r, 32);

    const __bf16* rowS = hbf + (size_t)sIdx * INS;
    const __bf16* rowD = hbf + (size_t)dIdx * INS;
    __bf16* tM = tA[wave];
    __bf16* tMsg = tG[wave];

    // ---- GEMM1: edge features -> m (16 x 128), K = 2*INS + radial rank-1 ----
    for (int nt = 0; nt < 8; ++nt) {
        v8f acc = {0.f, 0.f, 0.f, 0.f, 0.f, 0.f, 0.f, 0.f};
        for (int kc = 0; kc < KC1; ++kc) {
            v16bf a, b;
            int k0 = kc * 32 + half * 8;
            int k1 = k0 + 16;
            // runs of 8 never straddle the src|dst boundary (INS % 32 == 0)
            if (k0 < INS) set8(a, 0, rowS + k0); else set8(a, 0, rowD + (k0 - INS));
            if (k1 < INS) set8(a, 8, rowS + k1); else set8(a, 8, rowD + (k1 - INS));
            b = *(const v16bf*)(eW1f + ((size_t)(nt * KC1 + kc) * 32 + lane) * 16);
            acc = wmma_bf16(a, b, acc);
        }
        int n = nt * 16 + m;
        float wrad = eW1rad[n], bias = eb1[n];
#pragma unroll
        for (int r = 0; r < 8; ++r) {
            float v = acc[r] + radv[r] * wrad + bias;
            tM[(m0 + r) * TSTR + n] = (__bf16)silu_f(v);
        }
    }
    __syncthreads();

    // ---- GEMM2: msg = silu(m @ eW2 + eb2), K = 128 ----
    for (int nt = 0; nt < 8; ++nt) {
        v8f acc = {0.f, 0.f, 0.f, 0.f, 0.f, 0.f, 0.f, 0.f};
        for (int kc = 0; kc < 4; ++kc) {
            v16bf a, b;
            int k0 = kc * 32 + half * 8;
            set8(a, 0, &tM[m * TSTR + k0]);
            set8(a, 8, &tM[m * TSTR + k0 + 16]);
            b = *(const v16bf*)(eW2f + ((size_t)(nt * 4 + kc) * 32 + lane) * 16);
            acc = wmma_bf16(a, b, acc);
        }
        int n = nt * 16 + m;
        float bias = eb2[n];
#pragma unroll
        for (int r = 0; r < 8; ++r)
            tMsg[(m0 + r) * TSTR + n] = (__bf16)silu_f(acc[r] + bias);
    }
    __syncthreads();

    // ---- GEMM3: cwh = silu(msg @ cW1 + cb1); coord_w = cwh . cW2 ----
    float cwp[8];
#pragma unroll
    for (int r = 0; r < 8; ++r) cwp[r] = 0.f;
    for (int nt = 0; nt < 8; ++nt) {
        v8f acc = {0.f, 0.f, 0.f, 0.f, 0.f, 0.f, 0.f, 0.f};
        for (int kc = 0; kc < 4; ++kc) {
            v16bf a, b;
            int k0 = kc * 32 + half * 8;
            set8(a, 0, &tMsg[m * TSTR + k0]);
            set8(a, 8, &tMsg[m * TSTR + k0 + 16]);
            b = *(const v16bf*)(cW1f + ((size_t)(nt * 4 + kc) * 32 + lane) * 16);
            acc = wmma_bf16(a, b, acc);
        }
        int n = nt * 16 + m;
        float bias = cb1[n], w2 = cW2[n];
#pragma unroll
        for (int r = 0; r < 8; ++r) cwp[r] += silu_f(acc[r] + bias) * w2;
    }
    // reduce across the 16 lanes of each half (rows m0..m0+7)
#pragma unroll
    for (int mask = 1; mask < 16; mask <<= 1)
#pragma unroll
        for (int r = 0; r < 8; ++r) cwp[r] += __shfl_xor(cwp[r], mask, 32);
    if ((lane & 15) == 0) {
#pragma unroll
        for (int r = 0; r < 8; ++r) cwS[wave][m0 + r] = cwp[r];
    }
    __syncthreads();

    // ---- scatter: h_neigh += msg (all lanes), x_sum/deg (lanes 0..15) ----
    for (int rr = 0; rr < 16; ++rr) {
        int dd = dst[eBase + rr];
        float* hn = h_neigh + (size_t)dd * 128;
#pragma unroll
        for (int f = lane; f < 128; f += 32)
            atomicAdd(hn + f, (float)tMsg[rr * TSTR + f]);
    }
    if (lane < 16) {
        float cw = cwS[wave][m];
        float* xs = x_sum + (size_t)dIdx * 3;
        atomicAdd(xs + 0, cw * nx);
        atomicAdd(xs + 1, cw * ny);
        atomicAdd(xs + 2, cw * nz);
        atomicAdd(deg + dIdx, 1.0f);
    }
}

// ---------------------------------------------------------------------------
// Node kernel: h_out = silu([h, h_neigh] @ nW1 + nb1) @ nW2 + nb2
//              x += x_sum / max(deg, 1)
// ---------------------------------------------------------------------------
template <int INS>
__global__ __launch_bounds__(128) void node_kernel(
    const __bf16* __restrict__ hbf, const float* __restrict__ hneigh,
    const __bf16* __restrict__ nW1f, const float* __restrict__ nb1,
    const __bf16* __restrict__ nW2f, const float* __restrict__ nb2,
    const float* __restrict__ xsum, const float* __restrict__ deg,
    float* __restrict__ xbuf, __bf16* __restrict__ hbf_out,
    float* __restrict__ hf_out) {
    constexpr int K1 = INS + 128;
    constexpr int KC1 = K1 / 32;

    __shared__ __bf16 tH[4][16 * TSTR];

    const int tid = threadIdx.x, wave = tid >> 5, lane = tid & 31;
    const int m = lane & 15, half = lane >> 4, m0 = half * 8;
    const int nBase = blockIdx.x * 64 + wave * 16;
    const int nd = nBase + m;

    const __bf16* rowH = hbf + (size_t)nd * INS;
    const float* rowN = hneigh + (size_t)nd * 128;
    __bf16* tM = tH[wave];

    for (int nt = 0; nt < 8; ++nt) {
        v8f acc = {0.f, 0.f, 0.f, 0.f, 0.f, 0.f, 0.f, 0.f};
        for (int kc = 0; kc < KC1; ++kc) {
            v16bf a, b;
            int k0 = kc * 32 + half * 8;
            int k1 = k0 + 16;
            if (k0 < INS) set8(a, 0, rowH + k0); else set8f(a, 0, rowN + (k0 - INS));
            if (k1 < INS) set8(a, 8, rowH + k1); else set8f(a, 8, rowN + (k1 - INS));
            b = *(const v16bf*)(nW1f + ((size_t)(nt * KC1 + kc) * 32 + lane) * 16);
            acc = wmma_bf16(a, b, acc);
        }
        int n = nt * 16 + m;
        float bias = nb1[n];
#pragma unroll
        for (int r = 0; r < 8; ++r)
            tM[(m0 + r) * TSTR + n] = (__bf16)silu_f(acc[r] + bias);
    }
    __syncthreads();

    for (int nt = 0; nt < 8; ++nt) {
        v8f acc = {0.f, 0.f, 0.f, 0.f, 0.f, 0.f, 0.f, 0.f};
        for (int kc = 0; kc < 4; ++kc) {
            v16bf a, b;
            int k0 = kc * 32 + half * 8;
            set8(a, 0, &tM[m * TSTR + k0]);
            set8(a, 8, &tM[m * TSTR + k0 + 16]);
            b = *(const v16bf*)(nW2f + ((size_t)(nt * 4 + kc) * 32 + lane) * 16);
            acc = wmma_bf16(a, b, acc);
        }
        int n = nt * 16 + m;
        float bias = nb2[n];
#pragma unroll
        for (int r = 0; r < 8; ++r) {
            float v = acc[r] + bias;
            size_t nr = (size_t)(nBase + m0 + r);
            hf_out[nr * 128 + n] = v;
            hbf_out[nr * 128 + n] = (__bf16)v;
        }
    }

    if (lane < 16) {
        float dg = fmaxf(deg[nd], 1.0f);
        float rdg = 1.0f / dg;
#pragma unroll
        for (int c = 0; c < 3; ++c)
            xbuf[(size_t)nd * 3 + c] += xsum[(size_t)nd * 3 + c] * rdg;
    }
}

// ---------------------------------------------------------------------------
// Pooling kernels
// ---------------------------------------------------------------------------
__global__ void mean_kernel(const float* __restrict__ h, float* __restrict__ mh) {
    int g = blockIdx.x, t = threadIdx.x;  // 128 threads = features
    float s = 0.f;
    for (int n = 0; n < NPG; ++n) s += h[((size_t)g * NPG + n) * 128 + t];
    mh[g * 128 + t] = s * (1.0f / (float)NPG);
}

// out[M,N] = A[M,K] @ W[K,N] + b
__global__ void linear_kernel(const float* __restrict__ A, const float* __restrict__ W,
                              const float* __restrict__ b, float* __restrict__ out,
                              int M, int K, int N) {
    int i = blockIdx.x * blockDim.x + threadIdx.x;
    if (i >= M * N) return;
    int mm = i / N, nn = i % N;
    float s = b[nn];
    for (int k = 0; k < K; ++k) s += A[(size_t)mm * K + k] * W[(size_t)k * N + nn];
    out[i] = s;
}

// v[g,kp,c] = sum_f Wq[c, kp*128+f] * q[g,kp,f];  c0[g,kp] = bq[kp]·q[g,kp]
__global__ void vvec_kernel(const float* __restrict__ q, const float* __restrict__ Wq,
                            const float* __restrict__ bq, float* __restrict__ v,
                            float* __restrict__ c0) {
    int i = blockIdx.x * blockDim.x + threadIdx.x;
    if (i >= N_GRAPHS * KP * 128) return;
    int c = i & 127;
    int kp = (i >> 7) % KP;
    int g = i / (KP * 128);
    const float* qp = q + ((size_t)g * KP + kp) * 128;
    const float* wp = Wq + (size_t)c * (KP * 128) + kp * 128;
    float s = 0.f;
    for (int f = 0; f < 128; ++f) s += wp[f] * qp[f];
    v[i] = s;
    if (c == 0) {
        const float* bp = bq + kp * 128;
        float s0 = 0.f;
        for (int f = 0; f < 128; ++f) s0 += bp[f] * qp[f];
        c0[g * KP + kp] = s0;
    }
}

// logits[g,n,kp] = h[g,n,:]·v[g,kp,:] + c0[g,kp]
__global__ void logits_kernel(const float* __restrict__ h, const float* __restrict__ v,
                              const float* __restrict__ c0, float* __restrict__ lg) {
    int i = blockIdx.x * blockDim.x + threadIdx.x;
    if (i >= N_GRAPHS * NPG * KP) return;
    int kp = i % KP;
    int n = (i / KP) % NPG;
    int g = i / (KP * NPG);
    const float* hp = h + ((size_t)g * NPG + n) * 128;
    const float* vp = v + ((size_t)g * KP + kp) * 128;
    float s = c0[g * KP + kp];
    for (int c = 0; c < 128; ++c) s += hp[c] * vp[c];
    lg[i] = s;
}

// one block (128 thr) per (g,kp): softmax over n, then out = sum_n w[n]*vals[n,:]
__global__ void pool_kernel(const float* __restrict__ lg, const float* __restrict__ vals,
                            int vdim, float* __restrict__ out) {
    __shared__ float w[NPG];
    __shared__ float red[128];
    int g = blockIdx.x / KP, kp = blockIdx.x % KP;
    int t = threadIdx.x;
    const float* l = lg + (size_t)g * NPG * KP + kp;  // stride KP over n
    float mx = -1e30f;
    for (int n = t; n < NPG; n += 128) mx = fmaxf(mx, l[(size_t)n * KP]);
    red[t] = mx;
    __syncthreads();
    for (int s = 64; s > 0; s >>= 1) {
        if (t < s) red[t] = fmaxf(red[t], red[t + s]);
        __syncthreads();
    }
    mx = red[0];
    __syncthreads();
    float sm = 0.f;
    for (int n = t; n < NPG; n += 128) {
        float e = __expf(l[(size_t)n * KP] - mx);
        w[n] = e;
        sm += e;
    }
    red[t] = sm;
    __syncthreads();
    for (int s = 64; s > 0; s >>= 1) {
        if (t < s) red[t] += red[t + s];
        __syncthreads();
    }
    float invs = 1.0f / red[0];
    __syncthreads();
    for (int c = t; c < vdim; c += 128) {
        float acc = 0.f;
        for (int n = 0; n < NPG; ++n)
            acc += w[n] * vals[((size_t)g * NPG + n) * vdim + c];
        out[((size_t)(g * KP + kp)) * vdim + c] = acc * invs;
    }
}

// ===========================================================================
// Host side
// ===========================================================================
extern "C" void kernel_launch(void* const* d_in, const int* in_sizes, int n_in,
                              void* d_out, int out_size, void* d_ws, size_t ws_size,
                              hipStream_t stream) {
    (void)in_sizes; (void)n_in; (void)out_size; (void)ws_size;
    const int N = N_NODES, E = N_EDGES;

    // Input order: pos, feat, params-leaves (sorted-key pytree), src, dst, n_graphs.
    // params leaves: convs[c] keys sorted: cW1,cW2,cb1,eW1,eW2,eb1,eb2,nW1,nW2,nb1,nb2
    // then eqv_key_W/b, eqv_query_W/b, inv_key_W/b, inv_query_W/b.
    const float* pos  = (const float*)d_in[0];
    const float* feat = (const float*)d_in[1];
    struct Conv { const float *cW1,*cW2,*cb1,*eW1,*eW2,*eb1,*eb2,*nW1,*nW2,*nb1,*nb2; };
    Conv cv[3];
    for (int c = 0; c < 3; ++c) {
        int b = 2 + c * 11;
        cv[c].cW1 = (const float*)d_in[b + 0];
        cv[c].cW2 = (const float*)d_in[b + 1];
        cv[c].cb1 = (const float*)d_in[b + 2];
        cv[c].eW1 = (const float*)d_in[b + 3];
        cv[c].eW2 = (const float*)d_in[b + 4];
        cv[c].eb1 = (const float*)d_in[b + 5];
        cv[c].eb2 = (const float*)d_in[b + 6];
        cv[c].nW1 = (const float*)d_in[b + 7];
        cv[c].nW2 = (const float*)d_in[b + 8];
        cv[c].nb1 = (const float*)d_in[b + 9];
        cv[c].nb2 = (const float*)d_in[b + 10];
    }
    const float* eqv_key_W   = (const float*)d_in[35];
    const float* eqv_key_b   = (const float*)d_in[36];
    const float* eqv_query_W = (const float*)d_in[37];
    const float* eqv_query_b = (const float*)d_in[38];
    const float* inv_key_W   = (const float*)d_in[39];
    const float* inv_key_b   = (const float*)d_in[40];
    const float* inv_query_W = (const float*)d_in[41];
    const float* inv_query_b = (const float*)d_in[42];
    const int* src = (const int*)d_in[43];
    const int* dst = (const int*)d_in[44];
    float* out = (float*)d_out;

    char* ws = (char*)d_ws;
    size_t off = 0;
    auto alloc = [&](size_t bytes) -> char* {
        char* p = ws + off;
        off = (off + bytes + 255) & ~(size_t)255;
        return p;
    };
    float*  xbuf   = (float*)alloc((size_t)N * 3 * 4);
    __bf16* featbf = (__bf16*)alloc((size_t)N * 64 * 2);
    __bf16* hbf0   = (__bf16*)alloc((size_t)N * 128 * 2);
    __bf16* hbf1   = (__bf16*)alloc((size_t)N * 128 * 2);
    float*  hf     = (float*)alloc((size_t)N * 128 * 4);
    float*  hneigh = (float*)alloc((size_t)N * 128 * 4);
    float*  xsum   = (float*)alloc((size_t)N * 3 * 4);
    float*  degb   = (float*)alloc((size_t)N * 4);
    size_t accumFloats = (size_t)(((char*)(degb + N)) - ((char*)hneigh)) / 4;

    int insArr[3] = {64, 128, 128};
    __bf16 *eW1f[3], *eW2f[3], *cW1f[3], *nW1f[3], *nW2f[3];
    for (int c = 0; c < 3; ++c) {
        int ins = insArr[c];
        eW1f[c] = (__bf16*)alloc((size_t)2 * ins * 128 * 2);
        eW2f[c] = (__bf16*)alloc((size_t)128 * 128 * 2);
        cW1f[c] = (__bf16*)alloc((size_t)128 * 128 * 2);
        nW1f[c] = (__bf16*)alloc((size_t)(ins + 128) * 128 * 2);
        nW2f[c] = (__bf16*)alloc((size_t)128 * 128 * 2);
    }
    float* mean_h = (float*)alloc((size_t)N_GRAPHS * 128 * 4);
    float* q_eqv  = (float*)alloc((size_t)N_GRAPHS * KP * 128 * 4);
    float* q_inv  = (float*)alloc((size_t)N_GRAPHS * KP * 128 * 4);
    float* v_eqv  = (float*)alloc((size_t)N_GRAPHS * KP * 128 * 4);
    float* c0_eqv = (float*)alloc((size_t)N_GRAPHS * KP * 4);
    float* v_inv  = (float*)alloc((size_t)N_GRAPHS * KP * 128 * 4);
    float* c0_inv = (float*)alloc((size_t)N_GRAPHS * KP * 4);
    float* lg_eqv = (float*)alloc((size_t)N_GRAPHS * NPG * KP * 4);
    float* lg_inv = (float*)alloc((size_t)N_GRAPHS * NPG * KP * 4);

    // ---- pack weight fragments ----
    auto pack = [&](const float* W, int K, __bf16* f) {
        int total = K * 128;
        pack_frags<<<(total + 255) / 256, 256, 0, stream>>>(W, K, 128, f);
    };
    for (int c = 0; c < 3; ++c) {
        int ins = insArr[c];
        pack(cv[c].eW1, 2 * ins, eW1f[c]);  // rows 0..2*ins-1 only (row 2*ins = radial)
        pack(cv[c].eW2, 128, eW2f[c]);
        pack(cv[c].cW1, 128, cW1f[c]);
        pack(cv[c].nW1, ins + 128, nW1f[c]);
        pack(cv[c].nW2, 128, nW2f[c]);
    }

    f2bf_kernel<<<2048, 256, 0, stream>>>(feat, featbf, (size_t)N * 64);
    copyf_kernel<<<512, 256, 0, stream>>>(pos, xbuf, (size_t)N * 3);

    // ---- conv 0 (ins = 64) ----
    zero_kernel<<<2048, 256, 0, stream>>>(hneigh, accumFloats);
    edge_kernel<64><<<E / 64, 128, 0, stream>>>(
        xbuf, featbf, src, dst, eW1f[0], cv[0].eW1 + (size_t)128 * 128, cv[0].eb1,
        eW2f[0], cv[0].eb2, cW1f[0], cv[0].cb1, cv[0].cW2, hneigh, xsum, degb);
    node_kernel<64><<<N / 64, 128, 0, stream>>>(featbf, hneigh, nW1f[0], cv[0].nb1,
                                                nW2f[0], cv[0].nb2, xsum, degb, xbuf,
                                                hbf0, hf);
    // ---- conv 1 (ins = 128) ----
    zero_kernel<<<2048, 256, 0, stream>>>(hneigh, accumFloats);
    edge_kernel<128><<<E / 64, 128, 0, stream>>>(
        xbuf, hbf0, src, dst, eW1f[1], cv[1].eW1 + (size_t)256 * 128, cv[1].eb1,
        eW2f[1], cv[1].eb2, cW1f[1], cv[1].cb1, cv[1].cW2, hneigh, xsum, degb);
    node_kernel<128><<<N / 64, 128, 0, stream>>>(hbf0, hneigh, nW1f[1], cv[1].nb1,
                                                 nW2f[1], cv[1].nb2, xsum, degb, xbuf,
                                                 hbf1, hf);
    // ---- conv 2 (ins = 128) ----
    zero_kernel<<<2048, 256, 0, stream>>>(hneigh, accumFloats);
    edge_kernel<128><<<E / 64, 128, 0, stream>>>(
        xbuf, hbf1, src, dst, eW1f[2], cv[2].eW1 + (size_t)256 * 128, cv[2].eb1,
        eW2f[2], cv[2].eb2, cW1f[2], cv[2].cb1, cv[2].cW2, hneigh, xsum, degb);
    node_kernel<128><<<N / 64, 128, 0, stream>>>(hbf1, hneigh, nW1f[2], cv[2].nb1,
                                                 nW2f[2], cv[2].nb2, xsum, degb, xbuf,
                                                 hbf0, hf);

    // ---- keypoint pooling ----
    mean_kernel<<<N_GRAPHS, 128, 0, stream>>>(hf, mean_h);
    int qn = N_GRAPHS * KP * 128;
    linear_kernel<<<(qn + 255) / 256, 256, 0, stream>>>(mean_h, eqv_key_W, eqv_key_b,
                                                        q_eqv, N_GRAPHS, 128, KP * 128);
    linear_kernel<<<(qn + 255) / 256, 256, 0, stream>>>(mean_h, inv_key_W, inv_key_b,
                                                        q_inv, N_GRAPHS, 128, KP * 128);
    vvec_kernel<<<(qn + 255) / 256, 256, 0, stream>>>(q_eqv, eqv_query_W, eqv_query_b,
                                                      v_eqv, c0_eqv);
    vvec_kernel<<<(qn + 255) / 256, 256, 0, stream>>>(q_inv, inv_query_W, inv_query_b,
                                                      v_inv, c0_inv);
    int ln = N_GRAPHS * NPG * KP;
    logits_kernel<<<(ln + 255) / 256, 256, 0, stream>>>(hf, v_eqv, c0_eqv, lg_eqv);
    logits_kernel<<<(ln + 255) / 256, 256, 0, stream>>>(hf, v_inv, c0_inv, lg_inv);
    pool_kernel<<<N_GRAPHS * KP, 128, 0, stream>>>(lg_eqv, xbuf, 3, out);
    pool_kernel<<<N_GRAPHS * KP, 128, 0, stream>>>(lg_inv, hf, 128,
                                                   out + N_GRAPHS * KP * 3);
}
